// MultiHeadAttention_15573551415484
// MI455X (gfx1250) — matrix operand order repaired and checked by reading, not compiled, not run on hardware
//
#include <hip/hip_runtime.h>
#include <math.h>

// ---------------- problem constants ----------------
constexpr int B  = 4;
constexpr int S  = 2048;
constexpr int D  = 1024;
constexpr int H  = 16;
constexpr int DK = 64;              // == DV
constexpr int ST = S / 16;          // q-tiles per (b,h)
constexpr int MW = S / 32;          // mask words per row
constexpr float SCALE = 0.125f;     // 1/sqrt(64)

#ifndef __has_builtin
#define __has_builtin(x) 0
#endif
#if __has_builtin(__builtin_amdgcn_global_load_async_to_lds_b128)
#define HAS_ASYNC_LDS 1
#else
#define HAS_ASYNC_LDS 0
#endif

// ---------------- WMMA types ----------------
typedef __attribute__((ext_vector_type(16))) __bf16 v16bf;
typedef __attribute__((ext_vector_type(8)))  float  v8f;
typedef __attribute__((ext_vector_type(4)))  int    v4i;

union FragBF { v16bf v; unsigned int d[8]; };

__device__ __forceinline__ v8f v8f_zero() {
  v8f z;
#pragma unroll
  for (int i = 0; i < 8; ++i) z[i] = 0.0f;
  return z;
}

__device__ __forceinline__ unsigned short f32_to_bf16(float f) {
  unsigned int x = __float_as_uint(f);
  x += 0x7FFFu + ((x >> 16) & 1u);        // round to nearest even
  return (unsigned short)(x >> 16);
}

// 16-byte global -> LDS copy: async path if available, sync fallback else.
#if HAS_ASYNC_LDS
__device__ __forceinline__ void cp16_g2l(unsigned short* dst_lds,
                                         const unsigned short* src_glb) {
  __builtin_amdgcn_global_load_async_to_lds_b128(
      (__attribute__((address_space(1))) v4i*)(src_glb),
      (__attribute__((address_space(3))) v4i*)(dst_lds), 0, 0);
}
#define ASYNC_WAIT() asm volatile("s_wait_asynccnt 0x0" ::: "memory")
#else
__device__ __forceinline__ void cp16_g2l(unsigned short* dst_lds,
                                         const unsigned short* src_glb) {
  *(uint4*)dst_lds = *(const uint4*)src_glb;   // global_load_b128 + ds_store_b128
}
#define ASYNC_WAIT() ((void)0)
#endif

// A fragment: 16x32 bf16, row-major source, row stride ld (elements).
// lane m = lane&15 holds row m; dword i holds K = 2i + 8*half + (i>=4 ? 8:0).
__device__ __forceinline__ v16bf load_a_frag(const unsigned short* base, int ld, int lane) {
  const int m = lane & 15, half = lane >> 4;
  FragBF f;
#pragma unroll
  for (int i = 0; i < 8; ++i) {
    const int ks = 2 * i + 8 * half + ((i >= 4) ? 8 : 0);
    f.d[i] = *(const unsigned int*)(base + (size_t)m * ld + ks);
  }
  return f.v;
}

// B fragment: 32x16 bf16 from TRANSPOSED storage srcT[n][k] (k contiguous,
// row stride ld). lane n = lane&15 holds column n; dword i holds K = 2i + 8*half + (i>=4 ? 8:0).
__device__ __forceinline__ v16bf load_b_frag(const unsigned short* baseT, int ld, int lane) {
  const int n = lane & 15, half = lane >> 4;
  FragBF f;
#pragma unroll
  for (int i = 0; i < 8; ++i) {
    const int ks = 2 * i + 8 * half + ((i >= 4) ? 8 : 0);
    f.d[i] = *(const unsigned int*)(baseT + (size_t)n * ld + ks);
  }
  return f.v;
}

__device__ __forceinline__ v8f wmma_bf16(v16bf a, v16bf b, v8f c) {
  return __builtin_amdgcn_wmma_f32_16x16x32_bf16(false, a, false, b, (short)0, c,
                                                 false, false);
}

// row reductions within a 16-lane half (C-layout rows live in one half)
__device__ __forceinline__ float rowmax16(float v) {
#pragma unroll
  for (int m = 1; m <= 8; m <<= 1) v = fmaxf(v, __shfl_xor(v, m, 32));
  return v;
}
__device__ __forceinline__ float rowsum16(float v) {
#pragma unroll
  for (int m = 1; m <= 8; m <<= 1) v += __shfl_xor(v, m, 32);
  return v;
}

// ---------------- kernel 1: x fp32 -> bf16 ----------------
__global__ void cvt_x_kernel(const float* __restrict__ x,
                             unsigned short* __restrict__ xb, int n) {
  int i = blockIdx.x * blockDim.x + threadIdx.x;
  if (i < n) xb[i] = f32_to_bf16(x[i]);
}

// ---------------- kernel 2: W[H,D,64] fp32 -> W^T[H,64,D] bf16 (x3) ----------------
__global__ void cvt_w_kernel(const float* __restrict__ wq, const float* __restrict__ wk,
                             const float* __restrict__ wv,
                             unsigned short* __restrict__ wqt, unsigned short* __restrict__ wkt,
                             unsigned short* __restrict__ wvt) {
  int i = blockIdx.x * blockDim.x + threadIdx.x;   // over H*D*DK
  if (i >= H * D * DK) return;
  int n = i % DK;
  int d = (i / DK) % D;
  int h = i / (DK * D);
  size_t src = (size_t)(h * D + d) * DK + n;
  size_t dst = ((size_t)h * DK + n) * D + d;
  wqt[dst] = f32_to_bf16(wq[src]);
  wkt[dst] = f32_to_bf16(wk[src]);
  wvt[dst] = f32_to_bf16(wv[src]);
}

// ---------------- kernel 2b: pack bool mask into bit-words [S][S/32] ----------------
__global__ void pack_mask_kernel(const unsigned char* __restrict__ msk,
                                 unsigned int* __restrict__ mbits) {
  int i = blockIdx.x * blockDim.x + threadIdx.x;   // over S*MW
  if (i >= S * MW) return;
  int r = i / MW, w = i % MW;
  const unsigned char* p = msk + (size_t)r * S + w * 32;
  unsigned int word = 0;
#pragma unroll
  for (int c = 0; c < 32; ++c) word |= (p[c] ? 1u : 0u) << c;
  mbits[i] = word;
}

// ---------------- kernel 3: fused QKV projection ----------------
// 8 waves/block share (b,h); W tiles double-buffered in LDS via async copies.
// q,k natural [B,H,S,64]; v transposed [B,H,64,S].
__global__ __launch_bounds__(256) void proj_kernel(
    const unsigned short* __restrict__ xb,
    const unsigned short* __restrict__ wqt, const unsigned short* __restrict__ wkt,
    const unsigned short* __restrict__ wvt,
    unsigned short* __restrict__ q, unsigned short* __restrict__ k,
    unsigned short* __restrict__ vt) {
  __shared__ unsigned short wbuf[2][3][64][32];   // 24 KB, [mat][dk-row][k-chunk]

  const int tid  = threadIdx.x;
  const int lane = tid & 31;
  const int wgl  = blockIdx.x * 8 + (tid >> 5);
  const int bh = wgl / ST, st = wgl % ST;
  const int b = bh / H, h = bh % H;

  const unsigned short* xtile = xb + ((size_t)b * S + st * 16) * D;
  const unsigned short* wts[3] = {wqt, wkt, wvt};

  // cooperative stage of one 32-wide k-slab of all 3 weight matrices
  auto stage = [&](int buf, int kk) {
#pragma unroll
    for (int c = 0; c < 3; ++c) {
      const int qd  = tid + c * 256;      // chunk 0..767 (16B each)
      const int mat = qd >> 8;
      const int rem = qd & 255;
      const int row = rem >> 2, part = rem & 3;
      cp16_g2l(&wbuf[buf][mat][row][part * 8],
               wts[mat] + ((size_t)h * DK + row) * D + kk + part * 8);
    }
  };

  v8f acc[3][4];
#pragma unroll
  for (int m3 = 0; m3 < 3; ++m3)
#pragma unroll
    for (int t = 0; t < 4; ++t) acc[m3][t] = v8f_zero();

  stage(0, 0);
  ASYNC_WAIT();
  __syncthreads();

  for (int kk = 0; kk < D; kk += 32) {
    const int cur = (kk >> 5) & 1;
    if (kk + 32 < D) stage(cur ^ 1, kk + 32);

    v16bf a = load_a_frag(xtile + kk, D, lane);
#pragma unroll
    for (int m3 = 0; m3 < 3; ++m3) {
#pragma unroll
      for (int t = 0; t < 4; ++t) {
        v16bf bf = load_b_frag(&wbuf[cur][m3][t * 16][0], 32, lane);
        acc[m3][t] = wmma_bf16(a, bf, acc[m3][t]);
      }
    }
    ASYNC_WAIT();
    __syncthreads();
  }

  const int n = lane & 15, half = lane >> 4;
  const size_t qbase = ((size_t)(b * H + h) * S + st * 16) * DK;
  const size_t vbase = (size_t)(b * H + h) * DK * S;
#pragma unroll
  for (int t = 0; t < 4; ++t) {
#pragma unroll
    for (int j = 0; j < 8; ++j) {
      const int r  = j + 8 * half;
      const int ng = t * 16 + n;
      q[qbase + (size_t)r * DK + ng] = f32_to_bf16(acc[0][t][j]);
      k[qbase + (size_t)r * DK + ng] = f32_to_bf16(acc[1][t][j]);
      vt[vbase + (size_t)ng * S + st * 16 + r] = f32_to_bf16(acc[2][t][j]);
    }
  }
}

// ---------------- kernel 4: flash attention ----------------
// 8 waves/block share (b,h); K/V s-tiles double-buffered in LDS via async copies.
__global__ __launch_bounds__(256) void attn_kernel(
    const unsigned short* __restrict__ qp, const unsigned short* __restrict__ kp,
    const unsigned short* __restrict__ vtp, const unsigned int* __restrict__ mbits,
    float* __restrict__ out) {
  __shared__ unsigned short kbuf[2][32][64];   // 8 KB  [s-row][dk]
  __shared__ unsigned short vbuf[2][64][32];   // 8 KB  [dv][s-col]
  __shared__ unsigned short pbuf[8][16][32];   // 8 KB  per-wave P staging

  const int tid  = threadIdx.x;
  const int lane = tid & 31;
  const int w    = tid >> 5;
  const int wgl  = blockIdx.x * 8 + w;
  const int bh = wgl / ST, st = wgl % ST;
  const int b = bh / H, h = bh % H;
  const int q0 = st * 16;
  const int n = lane & 15, half = lane >> 4;

  const unsigned short* qb = qp + ((size_t)(b * H + h) * S + q0) * DK;
  const unsigned short* kb = kp + (size_t)(b * H + h) * S * DK;
  const unsigned short* vb = vtp + (size_t)(b * H + h) * DK * S;

  // cooperative stage of one 32-row K tile + matching V tile
  auto stage = [&](int buf, int s0) {
    cp16_g2l(&kbuf[buf][0][0] + tid * 8, kb + (size_t)s0 * DK + tid * 8);  // contiguous
    const int row = tid >> 2, part = tid & 3;
    cp16_g2l(&vbuf[buf][row][part * 8], vb + (size_t)row * S + s0 + part * 8);
  };

  const v16bf aq0 = load_a_frag(qb, DK, lane);
  const v16bf aq1 = load_a_frag(qb + 32, DK, lane);

  float mrow[8], lrow[8];
  v8f acc[4];
#pragma unroll
  for (int j = 0; j < 8; ++j) { mrow[j] = -__builtin_inff(); lrow[j] = 0.0f; }
#pragma unroll
  for (int t = 0; t < 4; ++t) acc[t] = v8f_zero();

  stage(0, 0);
  ASYNC_WAIT();
  __syncthreads();

  for (int s0 = 0; s0 < S; s0 += 32) {
    const int cur = (s0 >> 5) & 1;
    if (s0 + 32 < S) stage(cur ^ 1, s0 + 32);

    // ---- scores: two 16-col subtiles, K-dim 64 split in two WMMAs ----
    float p[2][8];
    unsigned int mword[8];
#pragma unroll
    for (int j = 0; j < 8; ++j)
      mword[j] = mbits[(size_t)(q0 + j + 8 * half) * MW + (s0 >> 5)];
#pragma unroll
    for (int u = 0; u < 2; ++u) {
      v8f sc = v8f_zero();
      sc = wmma_bf16(aq0, load_b_frag(&kbuf[cur][u * 16][0], DK, lane), sc);
      sc = wmma_bf16(aq1, load_b_frag(&kbuf[cur][u * 16][0] + 32, DK, lane), sc);
      const int cl = u * 16 + n;
#pragma unroll
      for (int j = 0; j < 8; ++j) {
        float v = sc[j] * SCALE;
        if ((mword[j] >> cl) & 1u) v = -__builtin_inff();
        p[u][j] = v;
      }
    }
    // ---- online softmax per row (rows live in 16-lane halves) ----
#pragma unroll
    for (int j = 0; j < 8; ++j) {
      float mx = rowmax16(fmaxf(p[0][j], p[1][j]));
      float mnew = fmaxf(mrow[j], mx);
      float e0 = __expf(p[0][j] - mnew);
      float e1 = __expf(p[1][j] - mnew);
      float rs = rowsum16(e0 + e1);
      float corr = __expf(mrow[j] - mnew);
      lrow[j] = lrow[j] * corr + rs;
      mrow[j] = mnew;
#pragma unroll
      for (int t = 0; t < 4; ++t) acc[t][j] *= corr;
      pbuf[w][j + 8 * half][n]      = f32_to_bf16(e0);
      pbuf[w][j + 8 * half][n + 16] = f32_to_bf16(e1);
    }
    asm volatile("s_wait_dscnt 0x0" ::: "memory");   // our P stores before our ds loads
    // ---- P in A-fragment layout (16x32, ld=32) ----
    FragBF pf;
    {
      const int m = lane & 15;
#pragma unroll
      for (int i = 0; i < 8; ++i) {
        const int ks = 2 * i + 8 * half + ((i >= 4) ? 8 : 0);
        pf.d[i] = *(const unsigned int*)&pbuf[w][m][ks];
      }
    }
    // ---- PV ----
#pragma unroll
    for (int t = 0; t < 4; ++t) {
      v16bf bv = load_b_frag(&vbuf[cur][t * 16][0], 32, lane);
      acc[t] = wmma_bf16(pf.v, bv, acc[t]);
    }

    ASYNC_WAIT();
    __syncthreads();
  }

  // normalize and store fp32 output [B,S,H*DV]
#pragma unroll
  for (int t = 0; t < 4; ++t) {
#pragma unroll
    for (int j = 0; j < 8; ++j) {
      const int r  = q0 + j + 8 * half;
      const int ng = t * 16 + n;
      out[((size_t)b * S + r) * (H * DK) + h * DK + ng] = acc[t][j] / lrow[j];
    }
  }
}

// ---------------- host launch ----------------
extern "C" void kernel_launch(void* const* d_in, const int* in_sizes, int n_in,
                              void* d_out, int out_size, void* d_ws, size_t ws_size,
                              hipStream_t stream) {
  (void)in_sizes; (void)n_in; (void)out_size; (void)ws_size;
  const float* x  = (const float*)d_in[0];
  const unsigned char* msk = (const unsigned char*)d_in[1];   // bool mask [S,S]
  const float* wq = (const float*)d_in[2];
  const float* wk = (const float*)d_in[3];
  const float* wv = (const float*)d_in[4];
  float* out = (float*)d_out;

  // workspace layout (bf16 as ushort)
  unsigned short* p = (unsigned short*)d_ws;
  unsigned short* xb  = p;                 p += (size_t)B * S * D;       // 16.8 MB
  unsigned short* wqt = p;                 p += (size_t)H * DK * D;      // 2.1 MB
  unsigned short* wkt = p;                 p += (size_t)H * DK * D;
  unsigned short* wvt = p;                 p += (size_t)H * DK * D;
  unsigned short* q   = p;                 p += (size_t)B * H * S * DK;  // 16.8 MB
  unsigned short* k   = p;                 p += (size_t)B * H * S * DK;
  unsigned short* vt  = p;                 p += (size_t)B * H * S * DK;  // [B,H,DK,S]
  unsigned int*   mb  = (unsigned int*)p;                                // 0.5 MB

  const int nx = B * S * D;
  cvt_x_kernel<<<(nx + 255) / 256, 256, 0, stream>>>(x, xb, nx);

  const int nw = H * D * DK;
  cvt_w_kernel<<<(nw + 255) / 256, 256, 0, stream>>>(wq, wk, wv, wqt, wkt, wvt);

  const int nm = S * MW;
  pack_mask_kernel<<<(nm + 255) / 256, 256, 0, stream>>>(msk, mb);

  const int waves = B * H * ST;            // 8192 waves, 8 per block
  proj_kernel<<<waves / 8, 256, 0, stream>>>(xb, wqt, wkt, wvt, q, k, vt);

  attn_kernel<<<waves / 8, 256, 0, stream>>>(q, k, vt, mb, out);
}
